// SelfAttentionWithPE_43611097923682
// MI455X (gfx1250) — matrix operand order repaired and checked by reading, not compiled, not run on hardware
//
#include <hip/hip_runtime.h>
#include <hip/hip_fp16.h>

// ---------------------------------------------------------------------------
// Self-attention with positional encoding, MI455X (gfx1250, wave32, WMMA).
// B=8, E=512, seq=1024, heads=8, head_dim=64. Compute-bound (~850 FLOP/B):
// all GEMMs and attention matmuls run on v_wmma_f32_16x16x32_f16 (f32 accum),
// with ping-pong software pipelining + global_prefetch_b8 hints.
// __launch_bounds__(256) keeps accumulators + both pipeline buffers in VGPRs
// (without it the default 1024-thread assumption caused scratch spills).
// ---------------------------------------------------------------------------

typedef __attribute__((ext_vector_type(16))) _Float16 v16h;
typedef __attribute__((ext_vector_type(8)))  _Float16 v8h;
typedef __attribute__((ext_vector_type(8)))  float    v8f;

#define E_DIM   512
#define SEQ     1024
#define NBATCH  8
#define HEADS   8
#define HD      64
#define M_TOTAL (NBATCH * SEQ)   // 8192 rows for the big GEMMs

__device__ __forceinline__ v8f wmma_f16(v16h a, v16h b, v8f c) {
  // 8 args: (neg_a, A, neg_b, B, c_mod, C, reuse_a, reuse_b)
  return __builtin_amdgcn_wmma_f32_16x16x32_f16(false, a, false, b,
                                                (short)0, c, false, false);
}

// A-fragment (16x32, f16), row-major source with leading dim `ld`.
// Layout: lanes 0-15 -> row m0+lane, K k0+0..15;
//         lanes 16-31 -> row m0+lane-16, K k0+16..31.
__device__ __forceinline__ v16h load_a_frag(const _Float16* base, int ld,
                                            int m0, int k0, int lane) {
  const int hs = lane >> 4, r = lane & 15;
  const _Float16* p = base + (size_t)(m0 + r) * ld + k0 + hs * 16;
  v8h lo = *(const v8h*)(p);        // K k0+hs*16   .. +7
  v8h hi = *(const v8h*)(p + 8);    // K k0+hs*16+8 .. +15
  v16h out;
  #pragma unroll
  for (int i = 0; i < 8; ++i) { out[i] = lo[i]; out[i + 8] = hi[i]; }
  return out;
}

// B-fragment (32x16, f16) from a "column-contiguous" matrix Bt[N][K]:
// lane holds column n0+(lane&15); contraction k0 + (lane<16 ? 0..15 : 16..31).
__device__ __forceinline__ v16h load_b_frag(const _Float16* bt, int ld,
                                            int k0, int n0, int lane) {
  const int hs = lane >> 4, r = lane & 15;
  const _Float16* p = bt + (size_t)(n0 + r) * ld + k0 + hs * 16;
  v8h lo = *(const v8h*)(p);
  v8h hi = *(const v8h*)(p + 8);
  v16h out;
  #pragma unroll
  for (int i = 0; i < 8; ++i) { out[i] = lo[i]; out[i + 8] = hi[i]; }
  return out;
}

// ---------------------------------------------------------------------------
// Kernel 1: xt[b,s,c] = x[b,c,r,w] + r*pos_w[0,c] + w*pos_w[1,c] + pos_b[c]
// ---------------------------------------------------------------------------
__global__ void prep_xt_kernel(const float* __restrict__ x,
                               const float* __restrict__ pos_w,
                               const float* __restrict__ pos_b,
                               _Float16* __restrict__ xt) {
  const size_t idx = (size_t)blockIdx.x * blockDim.x + threadIdx.x;
  const int c  = (int)(idx & 511);
  const int s  = (int)((idx >> 9) & 1023);
  const int b  = (int)(idx >> 19);
  const int rr = s >> 5, ww = s & 31;
  const float v = x[(((size_t)b * E_DIM + c) * 32 + rr) * 32 + ww]
                + (float)rr * pos_w[c] + (float)ww * pos_w[E_DIM + c] + pos_b[c];
  xt[idx] = (_Float16)v;
}

// ---------------------------------------------------------------------------
// Kernel 2: wt[n][k] = (f16) w[k][n]  (transpose so B-fragments are contiguous)
// ---------------------------------------------------------------------------
__global__ void wconv_kernel(const float* __restrict__ w,
                             _Float16* __restrict__ wt) {
  const int idx = blockIdx.x * blockDim.x + threadIdx.x;  // n*512 + k
  const int n = idx >> 9, kk = idx & 511;
  wt[idx] = (_Float16)w[kk * E_DIM + n];
}

// ---------------------------------------------------------------------------
// Kernel 3: C[8192,512] = A[8192,512] @ Wt^T + bias, fused scale & re-layout.
// mode 0: Q  -> f16 [b,h,s,d]  (scale = 1/sqrt(64) folded in)
// mode 1: K  -> f16 [b,h,s,d]
// mode 2: V  -> f16 [b,h,d,s]  (transposed for contiguous PV B-fragments)
// mode 3: out-> f32 [b,c,s]    (final output layout, bias = bo)
// Block = 256 threads (8 waves); each wave computes a 32x64 strip with
// ping-pong double buffering: buffer X computes while buffer Y loads.
// ---------------------------------------------------------------------------
__global__ void __launch_bounds__(256)
gemm512_kernel(const _Float16* __restrict__ A,
               const _Float16* __restrict__ Bt,
               const float* __restrict__ bias,
               void* __restrict__ outp,
               int mode, float scale) {
  const int lane = threadIdx.x & 31;
  const int wave = threadIdx.x >> 5;
  const int hs = lane >> 4, r = lane & 15;
  const int m0 = blockIdx.x * 32;
  const int n0 = wave * 64;

  const v8f z = {};
  v8f acc[2][4] = {{z, z, z, z}, {z, z, z, z}};

  // prologue: K-step 0 fragments into buffer X
  v16h a0 = load_a_frag(A, E_DIM, m0,      0, lane);
  v16h a1 = load_a_frag(A, E_DIM, m0 + 16, 0, lane);
  v16h bf[4];
  #pragma unroll
  for (int t = 0; t < 4; ++t) bf[t] = load_b_frag(Bt, E_DIM, 0, n0 + t * 16, lane);

  for (int kk = 0; kk < E_DIM / 32; kk += 2) {
    const int k1 = (kk + 1) * 32;
    const int k2 = (kk + 2) * 32;
    // prefetch ahead (lowers to global_prefetch_b8)
    if (k2 < E_DIM) {
      __builtin_prefetch(A  + (size_t)(m0 + r)      * E_DIM + k2, 0, 3);
      __builtin_prefetch(A  + (size_t)(m0 + 16 + r) * E_DIM + k2, 0, 3);
      __builtin_prefetch(Bt + (size_t)(n0 + r)      * E_DIM + k2, 0, 3);
      __builtin_prefetch(Bt + (size_t)(n0 + 32 + r) * E_DIM + k2, 0, 3);
    }
    // buffer Y loads K-step kk+1 while buffer X computes K-step kk
    v16h c0 = load_a_frag(A, E_DIM, m0,      k1, lane);
    v16h c1 = load_a_frag(A, E_DIM, m0 + 16, k1, lane);
    v16h cb[4];
    #pragma unroll
    for (int t = 0; t < 4; ++t) cb[t] = load_b_frag(Bt, E_DIM, k1, n0 + t * 16, lane);

    #pragma unroll
    for (int t = 0; t < 4; ++t) {
      acc[0][t] = wmma_f16(a0, bf[t], acc[0][t]);
      acc[1][t] = wmma_f16(a1, bf[t], acc[1][t]);
    }

    // buffer X loads K-step kk+2 while buffer Y computes K-step kk+1
    if (k2 < E_DIM) {
      a0 = load_a_frag(A, E_DIM, m0,      k2, lane);
      a1 = load_a_frag(A, E_DIM, m0 + 16, k2, lane);
      #pragma unroll
      for (int t = 0; t < 4; ++t) bf[t] = load_b_frag(Bt, E_DIM, k2, n0 + t * 16, lane);
    }

    #pragma unroll
    for (int t = 0; t < 4; ++t) {
      acc[0][t] = wmma_f16(c0, cb[t], acc[0][t]);
      acc[1][t] = wmma_f16(c1, cb[t], acc[1][t]);
    }
  }

  #pragma unroll
  for (int rt = 0; rt < 2; ++rt) {
    #pragma unroll
    for (int t = 0; t < 4; ++t) {
      const int n = n0 + t * 16 + r;
      const float bb = bias[n];
      #pragma unroll
      for (int i = 0; i < 8; ++i) {
        const int row  = m0 + rt * 16 + i + hs * 8;
        const int bidx = row >> 10;      // batch
        const int s    = row & 1023;     // sequence position
        const float val = (acc[rt][t][i] + bb) * scale;
        if (mode <= 1) {              // Q or K: [b,h,s,d] f16
          const int hh = n >> 6, d = n & 63;
          ((_Float16*)outp)[(((size_t)bidx * HEADS + hh) * SEQ + s) * HD + d] =
              (_Float16)val;
        } else if (mode == 2) {       // V transposed: [b,h,d,s] f16
          const int hh = n >> 6, d = n & 63;
          ((_Float16*)outp)[(((size_t)bidx * HEADS + hh) * HD + d) * SEQ + s] =
              (_Float16)val;
        } else {                      // final output: [b, c, s] f32
          ((float*)outp)[((size_t)bidx * E_DIM + n) * SEQ + s] = val;
        }
      }
    }
  }
}

// ---------------------------------------------------------------------------
// Kernel 4: flash attention. grid = (b*h, seq/128), block = 256 (8 waves).
// Each wave: 16 query rows, streams 32 keys/iter with online softmax.
// Pipelining: after the S-tile WMMAs, the *next* tile's K fragments and this
// tile's V fragments are loaded so their latency overlaps the softmax VALU.
// Row statistics live in 16-lane groups (C-fragment layout), reduced with
// __shfl_xor width 16 (wave32). P goes through LDS to become an A-fragment.
// ---------------------------------------------------------------------------
__global__ void __launch_bounds__(256)
attn_kernel(const _Float16* __restrict__ q,
            const _Float16* __restrict__ k,
            const _Float16* __restrict__ vt,
            _Float16* __restrict__ attn_out) {
  __shared__ __align__(16) _Float16 pbuf[8 * 16 * 32];  // 8 KiB, 1 tile/wave
  const int lane = threadIdx.x & 31;
  const int wave = threadIdx.x >> 5;
  const int hs = lane >> 4, r = lane & 15;
  const int bh = blockIdx.x;               // b*8 + h
  const int b = bh >> 3, h = bh & 7;
  const int q0 = blockIdx.y * 128 + wave * 16;

  const _Float16* qb = q  + (size_t)bh * SEQ * HD;   // [s,d], scale pre-folded
  const _Float16* kb = k  + (size_t)bh * SEQ * HD;   // [s,d] == Bt for QK^T
  const _Float16* vb = vt + (size_t)bh * HD * SEQ;   // [d,s] == Bt for P@V
  _Float16* lp = &pbuf[wave * 16 * 32];

  const v16h aq0 = load_a_frag(qb, HD, q0, 0,  lane);
  const v16h aq1 = load_a_frag(qb, HD, q0, 32, lane);

  float mrow[8], lrow[8];
  const v8f z = {};
  v8f acc[4] = {z, z, z, z};
  #pragma unroll
  for (int i = 0; i < 8; ++i) { mrow[i] = -1e30f; lrow[i] = 0.0f; }

  // prologue: K fragments for key tile 0
  v16h bk[4];
  bk[0] = load_b_frag(kb, HD, 0,  0,  lane);
  bk[1] = load_b_frag(kb, HD, 32, 0,  lane);
  bk[2] = load_b_frag(kb, HD, 0,  16, lane);
  bk[3] = load_b_frag(kb, HD, 32, 16, lane);

  for (int kt = 0; kt < SEQ / 32; ++kt) {
    const int key0 = kt * 32;

    // S tiles (16q x 32 keys) from current K fragments
    v8f s0 = {}, s1 = {};
    s0 = wmma_f16(aq0, bk[0], s0);
    s0 = wmma_f16(aq1, bk[1], s0);
    s1 = wmma_f16(aq0, bk[2], s1);
    s1 = wmma_f16(aq1, bk[3], s1);

    // start next tile's K loads + this tile's V loads (overlap with softmax)
    const int nkey = key0 + 32;
    if (nkey < SEQ) {
      __builtin_prefetch(kb + (size_t)(nkey + 32 + r) * HD, 0, 3);
      bk[0] = load_b_frag(kb, HD, 0,  nkey,      lane);
      bk[1] = load_b_frag(kb, HD, 32, nkey,      lane);
      bk[2] = load_b_frag(kb, HD, 0,  nkey + 16, lane);
      bk[3] = load_b_frag(kb, HD, 32, nkey + 16, lane);
    }
    v16h bv[4];
    #pragma unroll
    for (int t4 = 0; t4 < 4; ++t4)
      bv[t4] = load_b_frag(vb, SEQ, key0, t4 * 16, lane);

    // online softmax over the 32-key slab
    #pragma unroll
    for (int i = 0; i < 8; ++i) {
      float v0 = s0[i], v1 = s1[i];
      float t = fmaxf(v0, v1);                       // tile row-max
      t = fmaxf(t, __shfl_xor(t, 1, 16));
      t = fmaxf(t, __shfl_xor(t, 2, 16));
      t = fmaxf(t, __shfl_xor(t, 4, 16));
      t = fmaxf(t, __shfl_xor(t, 8, 16));
      const float nm    = fmaxf(mrow[i], t);
      const float alpha = __expf(mrow[i] - nm);
      const float p0 = __expf(v0 - nm);
      const float p1 = __expf(v1 - nm);
      float ps = p0 + p1;                            // tile row-sum
      ps += __shfl_xor(ps, 1, 16);
      ps += __shfl_xor(ps, 2, 16);
      ps += __shfl_xor(ps, 4, 16);
      ps += __shfl_xor(ps, 8, 16);
      lrow[i] = lrow[i] * alpha + ps;
      mrow[i] = nm;
      #pragma unroll
      for (int t4 = 0; t4 < 4; ++t4) acc[t4][i] *= alpha;
      const int row = i + hs * 8;                    // C-frag -> LDS (16x32)
      lp[row * 32 + r]      = (_Float16)p0;
      lp[row * 32 + 16 + r] = (_Float16)p1;
    }

    // Re-load P as an A-fragment (ds traffic; same-wave DS ops are in-order).
    const v16h pa = load_a_frag(lp, 32, 0, 0, lane);
    #pragma unroll
    for (int t4 = 0; t4 < 4; ++t4)
      acc[t4] = wmma_f16(pa, bv[t4], acc[t4]);
  }

  float inv[8];
  #pragma unroll
  for (int i = 0; i < 8; ++i) inv[i] = 1.0f / lrow[i];
  #pragma unroll
  for (int t4 = 0; t4 < 4; ++t4) {
    #pragma unroll
    for (int i = 0; i < 8; ++i) {
      const int srow = q0 + i + hs * 8;
      const int col  = h * HD + t4 * 16 + r;
      attn_out[((size_t)b * SEQ + srow) * E_DIM + col] =
          (_Float16)(acc[t4][i] * inv[i]);
    }
  }
}

// ---------------------------------------------------------------------------
// Host-side orchestration (graph-capture safe: only kernel launches on stream)
// ---------------------------------------------------------------------------
extern "C" void kernel_launch(void* const* d_in, const int* in_sizes, int n_in,
                              void* d_out, int out_size, void* d_ws, size_t ws_size,
                              hipStream_t stream) {
  (void)in_sizes; (void)n_in; (void)out_size; (void)ws_size;
  const float* x     = (const float*)d_in[0];
  const float* pos_w = (const float*)d_in[1];
  const float* pos_b = (const float*)d_in[2];
  const float* wq = (const float*)d_in[3];
  const float* bq = (const float*)d_in[4];
  const float* wk = (const float*)d_in[5];
  const float* bk = (const float*)d_in[6];
  const float* wv = (const float*)d_in[7];
  const float* bv = (const float*)d_in[8];
  const float* wo = (const float*)d_in[9];
  const float* bo = (const float*)d_in[10];

  // Workspace carve-out (all f16): 5*8MiB activations + 4*0.5MiB weights = 42MiB
  const size_t ACT_N = (size_t)NBATCH * SEQ * E_DIM;  // 4,194,304 halfs
  const size_t W_N   = (size_t)E_DIM * E_DIM;         //   262,144 halfs
  _Float16* ws  = (_Float16*)d_ws;
  _Float16* xt  = ws;  ws += ACT_N;
  _Float16* wqT = ws;  ws += W_N;
  _Float16* wkT = ws;  ws += W_N;
  _Float16* wvT = ws;  ws += W_N;
  _Float16* woT = ws;  ws += W_N;
  _Float16* qbf = ws;  ws += ACT_N;   // [b,h,s,d]
  _Float16* kbf = ws;  ws += ACT_N;   // [b,h,s,d]
  _Float16* vtb = ws;  ws += ACT_N;   // [b,h,d,s]
  _Float16* atb = ws;  ws += ACT_N;   // [b,s,c]

  prep_xt_kernel<<<(int)(ACT_N / 256), 256, 0, stream>>>(x, pos_w, pos_b, xt);
  wconv_kernel<<<(int)(W_N / 256), 256, 0, stream>>>(wq, wqT);
  wconv_kernel<<<(int)(W_N / 256), 256, 0, stream>>>(wk, wkT);
  wconv_kernel<<<(int)(W_N / 256), 256, 0, stream>>>(wv, wvT);
  wconv_kernel<<<(int)(W_N / 256), 256, 0, stream>>>(wo, woT);

  gemm512_kernel<<<M_TOTAL / 32, 256, 0, stream>>>(xt, wqT, bq, qbf, 0, 0.125f);
  gemm512_kernel<<<M_TOTAL / 32, 256, 0, stream>>>(xt, wkT, bk, kbf, 1, 1.0f);
  gemm512_kernel<<<M_TOTAL / 32, 256, 0, stream>>>(xt, wvT, bv, vtb, 2, 1.0f);

  attn_kernel<<<dim3(NBATCH * HEADS, SEQ / 128), 256, 0, stream>>>(qbf, kbf, vtb, atb);

  gemm512_kernel<<<M_TOTAL / 32, 256, 0, stream>>>(atb, woT, bo, d_out, 3, 1.0f);
}
